// Inception_87522843558242
// MI455X (gfx1250) — 8-bit matrix operand order repaired and checked by reading, compiled, not run on hardware
//
#include <hip/hip_runtime.h>
#include <hip/hip_bf16.h>
#include <math.h>

typedef __attribute__((ext_vector_type(8))) int v8i;

#define EPSBN 1e-5f
#define BATCH 4096

#define OFF_HPAD   0L
#define SZ_HPAD    (4096L*18*18*16)
#define OFF_FC1IN  (OFF_HPAD + SZ_HPAD)
#define SZ_FC1IN   (4096L*4736)
#define OFF_FC1OUT (OFF_FC1IN + SZ_FC1IN)
#define SZ_FC1OUT  (4096L*1024)
#define OFF_WA     (OFF_FC1OUT + SZ_FC1OUT)
#define SZ_WA      (1L*2*32*8*4)
#define OFF_WB     (OFF_WA + SZ_WA)
#define SZ_WB      (3L*2*32*8*4)
#define OFF_WC     (OFF_WB + SZ_WB)
#define SZ_WC      (7L*2*32*8*4)
#define OFF_WF1    (OFF_WC + SZ_WC)
#define SZ_WF1     (74L*64*32*8*4)
#define OFF_WF2    (OFF_WF1 + SZ_WF1)
#define SZ_WF2     (10L*1024)

__global__ void k_zero(uint4* __restrict__ p, long n16) {
  long i = (long)blockIdx.x * blockDim.x + threadIdx.x;
  if (i < n16) { uint4 z; z.x = z.y = z.z = z.w = 0u; p[i] = z; }
}

__global__ void k_prep_convw(const float* __restrict__ w, int KH, int KW, int KSTEPS,
                             unsigned* __restrict__ wbuf) {
  int idx = blockIdx.x * blockDim.x + threadIdx.x;
  int total = KSTEPS * 2 * 32 * 8;
  if (idx >= total) return;
  int v    = idx & 7;
  int lane = (idx >> 3) & 31;
  int nt   = (idx >> 8) & 1;
  int ks   = idx >> 9;
  int l16  = lane >> 4;
  int kloc = (v >> 1) * 16 + l16 * 8 + (v & 1) * 4;
  int o    = nt * 16 + (lane & 15);
  int kk   = KH * KW * 16;
  unsigned dw = 0;
  for (int t = 0; t < 4; ++t) {
    int K = ks * 64 + kloc + t;
    int byte = 0;
    if (K < kk) {
      int c = K & 15, r = K >> 4;
      int dy = r / KW, dx = r % KW;
      float wv = w[((o * 16 + c) * KH + dy) * KW + dx];
      byte = (wv >= 0.f) ? 1 : 0xFF;
    }
    dw |= ((unsigned)(byte & 0xFF)) << (8 * t);
  }
  wbuf[idx] = dw;
}

__global__ void k_prep_fc1w(const float* __restrict__ w, unsigned* __restrict__ wbuf) {
  int idx = blockIdx.x * blockDim.x + threadIdx.x;
  if (idx >= 74 * 64 * 32 * 8) return;
  int v    = idx & 7;
  int lane = (idx >> 3) & 31;
  int nt   = (idx >> 8) & 63;
  int ks   = idx >> 14;
  int l16  = lane >> 4;
  int kloc = (v >> 1) * 16 + l16 * 8 + (v & 1) * 4;
  int o    = nt * 16 + (lane & 15);
  unsigned dw = 0;
  for (int t = 0; t < 4; ++t) {
    int K = ks * 64 + kloc + t;
    int byte = 0;
    if (K < 4704) byte = (w[(long)o * 4704 + K] >= 0.f) ? 1 : 0xFF;
    dw |= ((unsigned)(byte & 0xFF)) << (8 * t);
  }
  wbuf[idx] = dw;
}

__global__ void k_prep_fc2w(const float* __restrict__ w, signed char* __restrict__ out) {
  int i = blockIdx.x * blockDim.x + threadIdx.x;
  if (i < 10 * 1024) out[i] = (w[i] >= 0.f) ? 1 : -1;
}

__global__ void k_stem(const float* __restrict__ x,  const float* __restrict__ w1,
                       const float* __restrict__ b1, const float* __restrict__ g1,
                       const float* __restrict__ bt1,const float* __restrict__ m1,
                       const float* __restrict__ v1, signed char* __restrict__ hpad) {
  __shared__ float sw[144];
  __shared__ float sb[16];
  __shared__ float sth[16];
  int tid = threadIdx.x;
  if (tid < 144) sw[tid] = (w1[tid] >= 0.f) ? 1.f : -1.f;
  if (tid < 16) {
    sb[tid]  = b1[tid];
    sth[tid] = m1[tid] - bt1[tid] * sqrtf(v1[tid] + EPSBN) / g1[tid];
  }
  __syncthreads();
  int gid = blockIdx.x * blockDim.x + tid;
  if (gid >= BATCH * 196) return;
  int b = gid / 196, p = gid % 196;
  int py = p / 14, px = p % 14;
  float patch[4][4];
  const float* xb = x + (long)b * 784;
#pragma unroll
  for (int r = 0; r < 4; ++r) {
    int iy = 2 * py - 1 + r;
#pragma unroll
    for (int c = 0; c < 4; ++c) {
      int ix = 2 * px - 1 + c;
      patch[r][c] = (iy >= 0 && iy < 28 && ix >= 0 && ix < 28) ? xb[iy * 28 + ix] : 0.f;
    }
  }
  signed char* out = hpad + (long)b * 5184 + ((py + 2) * 18 + (px + 2)) * 16;
  for (int c = 0; c < 16; ++c) {
    float mx = -1e30f;
#pragma unroll
    for (int s = 0; s < 4; ++s) {
      int sy = s >> 1, sx = s & 1;
      float acc = 0.f;
#pragma unroll
      for (int dy = 0; dy < 3; ++dy)
#pragma unroll
        for (int dx = 0; dx < 3; ++dx)
          acc += patch[sy + dy][sx + dx] * sw[c * 9 + dy * 3 + dx];
      mx = fmaxf(mx, acc);
    }
    out[c] = (mx + sb[c] >= sth[c]) ? 1 : -1;
  }
}

template<int KH, int KW, int PAD, int KSTEPS>
__global__ void k_bconv(const signed char* __restrict__ hpad,
                        const v8i* __restrict__ wbuf,
                        const float* __restrict__ bias, const float* __restrict__ g,
                        const float* __restrict__ bt,   const float* __restrict__ bm,
                        const float* __restrict__ bv,
                        signed char* __restrict__ fc1in, int chan_off) {
  int wave = (blockIdx.x * blockDim.x + threadIdx.x) >> 5;
  int lane = threadIdx.x & 31;
  int m_base = wave * 16;
  int mrow = m_base + (lane & 15);
  int sub = mrow & 3;
  int pp  = mrow >> 2;
  int px  = pp % 7;
  int py  = (pp / 7) % 7;
  int b   = pp / 49;
  int y   = 2 * py + (sub >> 1);
  int xx  = 2 * px + (sub & 1);
  const signed char* hb = hpad + (long)b * 5184 + (lane >> 4) * 8;
  v8i c0 = {0, 0, 0, 0, 0, 0, 0, 0};
  v8i c1 = c0;
  for (int ks = 0; ks < KSTEPS; ++ks) {
    v8i a;
#pragma unroll
    for (int j = 0; j < 4; ++j) {
      int r = ks * 4 + j;
      if (r < KH * KW) {
        int dy = r / KW, dx = r % KW;
        int row = y + 2 - PAD + dy;
        int col = xx + 2 - PAD + dx;
        uint2 d = *reinterpret_cast<const uint2*>(hb + (row * 18 + col) * 16);
        a[2 * j] = (int)d.x; a[2 * j + 1] = (int)d.y;
      } else {
        a[2 * j] = 0; a[2 * j + 1] = 0;
      }
    }
    v8i w0 = wbuf[(ks * 2 + 0) * 32 + lane];
    v8i w1 = wbuf[(ks * 2 + 1) * 32 + lane];
    c0 = __builtin_amdgcn_wmma_i32_16x16x64_iu8(true, a, true, w0, c0, false, false);
    c1 = __builtin_amdgcn_wmma_i32_16x16x64_iu8(true, a, true, w1, c1, false, false);
  }
  int n  = lane & 15;
  int hh = lane >> 4;
  int q0 = (m_base + 8 * hh) >> 2;
#pragma unroll
  for (int nt = 0; nt < 2; ++nt) {
    int cl = nt * 16 + n;
    float bs  = bias[cl];
    float thr = bm[cl] - bt[cl] * sqrtf(bv[cl] + EPSBN) / g[cl];
    v8i acc = nt ? c1 : c0;
#pragma unroll
    for (int gq = 0; gq < 2; ++gq) {
      int a0 = acc[4 * gq], a1 = acc[4 * gq + 1], a2 = acc[4 * gq + 2], a3 = acc[4 * gq + 3];
      int mA = a0 > a1 ? a0 : a1;
      int mB = a2 > a3 ? a2 : a3;
      int mx = mA > mB ? mA : mB;
      signed char ov = ((float)mx + bs >= thr) ? 1 : -1;
      int q = q0 + gq;
      int bq = q / 49, rem = q % 49;
      fc1in[(long)bq * 4736 + (chan_off + cl) * 49 + rem] = ov;
    }
  }
}

__global__ void k_fc1(const signed char* __restrict__ A, const v8i* __restrict__ wbuf,
                      const float* __restrict__ bf1, const float* __restrict__ g,
                      const float* __restrict__ bt,  const float* __restrict__ bm,
                      const float* __restrict__ bv,  signed char* __restrict__ out) {
  int wave  = (blockIdx.x * blockDim.x + threadIdx.x) >> 5;
  int lane  = threadIdx.x & 31;
  int nquad = wave & 15;
  int tm    = wave >> 4;
  int ntb   = nquad * 4;
  const signed char* arow = A + (long)(tm * 16 + (lane & 15)) * 4736 + (lane >> 4) * 8;
  v8i c0 = {0, 0, 0, 0, 0, 0, 0, 0};
  v8i c1 = c0, c2 = c0, c3 = c0;
  for (int ks = 0; ks < 74; ++ks) {
    v8i a;
#pragma unroll
    for (int j = 0; j < 4; ++j) {
      uint2 d = *reinterpret_cast<const uint2*>(arow + ks * 64 + j * 16);
      a[2 * j] = (int)d.x; a[2 * j + 1] = (int)d.y;
    }
    const v8i* wp = wbuf + (long)(ks * 64 + ntb) * 32 + lane;
    if (ks + 1 < 74)
      __builtin_prefetch((const void*)(wp + 64 * 32), 0, 1);
    v8i w0 = wp[0];
    v8i w1 = wp[32];
    v8i w2 = wp[64];
    v8i w3 = wp[96];
    c0 = __builtin_amdgcn_wmma_i32_16x16x64_iu8(true, a, true, w0, c0, false, false);
    c1 = __builtin_amdgcn_wmma_i32_16x16x64_iu8(true, a, true, w1, c1, false, false);
    c2 = __builtin_amdgcn_wmma_i32_16x16x64_iu8(true, a, true, w2, c2, false, false);
    c3 = __builtin_amdgcn_wmma_i32_16x16x64_iu8(true, a, true, w3, c3, false, false);
  }
  int n = lane & 15, hh = lane >> 4;
#pragma unroll
  for (int nt = 0; nt < 4; ++nt) {
    int oc = (ntb + nt) * 16 + n;
    float bs  = bf1[oc];
    float thr = bm[oc] - bt[oc] * sqrtf(bv[oc] + EPSBN) / g[oc];
    v8i acc = (nt == 0) ? c0 : (nt == 1) ? c1 : (nt == 2) ? c2 : c3;
#pragma unroll
    for (int i = 0; i < 8; ++i) {
      int mm = tm * 16 + hh * 8 + i;
      out[(long)mm * 1024 + oc] = ((float)acc[i] + bs >= thr) ? 1 : -1;
    }
  }
}

static __device__ inline int dot4i8(int a, int b) {
  int s = 0;
#pragma unroll
  for (int t = 0; t < 4; ++t) {
    s += (int)(signed char)(a >> (8 * t)) * (int)(signed char)(b >> (8 * t));
  }
  return s;
}

__global__ void k_fc2(const signed char* __restrict__ X, const signed char* __restrict__ W,
                      const float* __restrict__ bf2, const float* __restrict__ g,
                      const float* __restrict__ bt,  const float* __restrict__ bm,
                      const float* __restrict__ bv,  float* __restrict__ out) {
  int b = blockIdx.x * blockDim.x + threadIdx.x;
  if (b >= BATCH) return;
  const int* xr = reinterpret_cast<const int*>(X + (long)b * 1024);
  int acc[10];
#pragma unroll
  for (int c = 0; c < 10; ++c) acc[c] = 0;
  for (int k = 0; k < 256; ++k) {
    int xv = xr[k];
#pragma unroll
    for (int c = 0; c < 10; ++c) {
      int wv = reinterpret_cast<const int*>(W + c * 1024)[k];
      acc[c] += dot4i8(xv, wv);
    }
  }
  float z[10];
  float mx = -1e30f;
#pragma unroll
  for (int c = 0; c < 10; ++c) {
    float y = (float)acc[c] + bf2[c];
    z[c] = (y - bm[c]) * (g[c] * rsqrtf(bv[c] + EPSBN)) + bt[c];
    mx = fmaxf(mx, z[c]);
  }
  float s = 0.f;
#pragma unroll
  for (int c = 0; c < 10; ++c) s += expf(z[c] - mx);
  float ls = logf(s);
#pragma unroll
  for (int c = 0; c < 10; ++c) out[b * 10 + c] = z[c] - mx - ls;
}

extern "C" void kernel_launch(void* const* d_in, const int* in_sizes, int n_in,
                              void* d_out, int out_size, void* d_ws, size_t ws_size,
                              hipStream_t stream) {
  (void)in_sizes; (void)n_in; (void)out_size; (void)ws_size;
  const float* x   = (const float*)d_in[0];
  const float* w1  = (const float*)d_in[1];
  const float* b1  = (const float*)d_in[2];
  const float* g1  = (const float*)d_in[3];
  const float* bt1 = (const float*)d_in[4];
  const float* m1  = (const float*)d_in[5];
  const float* v1  = (const float*)d_in[6];
  const float* wa  = (const float*)d_in[7];
  const float* ba  = (const float*)d_in[8];
  const float* ag  = (const float*)d_in[9];
  const float* abt = (const float*)d_in[10];
  const float* am  = (const float*)d_in[11];
  const float* av  = (const float*)d_in[12];
  const float* wb  = (const float*)d_in[13];
  const float* bb  = (const float*)d_in[14];
  const float* bg  = (const float*)d_in[15];
  const float* bbt = (const float*)d_in[16];
  const float* bm  = (const float*)d_in[17];
  const float* bv  = (const float*)d_in[18];
  const float* wc  = (const float*)d_in[19];
  const float* bc  = (const float*)d_in[20];
  const float* cg  = (const float*)d_in[21];
  const float* cbt = (const float*)d_in[22];
  const float* cm  = (const float*)d_in[23];
  const float* cv  = (const float*)d_in[24];
  const float* wf1 = (const float*)d_in[25];
  const float* bf1 = (const float*)d_in[26];
  const float* f1g = (const float*)d_in[27];
  const float* f1bt= (const float*)d_in[28];
  const float* f1m = (const float*)d_in[29];
  const float* f1v = (const float*)d_in[30];
  const float* wf2 = (const float*)d_in[31];
  const float* bf2 = (const float*)d_in[32];
  const float* f2g = (const float*)d_in[33];
  const float* f2bt= (const float*)d_in[34];
  const float* f2m = (const float*)d_in[35];
  const float* f2v = (const float*)d_in[36];

  char* ws = (char*)d_ws;
  signed char* hpad   = (signed char*)(ws + OFF_HPAD);
  signed char* fc1in  = (signed char*)(ws + OFF_FC1IN);
  signed char* fc1out = (signed char*)(ws + OFF_FC1OUT);
  unsigned* wa_p  = (unsigned*)(ws + OFF_WA);
  unsigned* wb_p  = (unsigned*)(ws + OFF_WB);
  unsigned* wc_p  = (unsigned*)(ws + OFF_WC);
  unsigned* wf1_p = (unsigned*)(ws + OFF_WF1);
  signed char* wf2_p = (signed char*)(ws + OFF_WF2);

  long n16 = (SZ_HPAD + SZ_FC1IN) / 16;
  k_zero<<<(int)((n16 + 255) / 256), 256, 0, stream>>>((uint4*)ws, n16);

  k_prep_convw<<<(1 * 2 * 32 * 8 + 255) / 256, 256, 0, stream>>>(wa, 1, 1, 1, wa_p);
  k_prep_convw<<<(3 * 2 * 32 * 8 + 255) / 256, 256, 0, stream>>>(wb, 3, 3, 3, wb_p);
  k_prep_convw<<<(7 * 2 * 32 * 8 + 255) / 256, 256, 0, stream>>>(wc, 5, 5, 7, wc_p);
  k_prep_fc1w<<<(74 * 64 * 32 * 8 + 255) / 256, 256, 0, stream>>>(wf1, wf1_p);
  k_prep_fc2w<<<(10 * 1024 + 255) / 256, 256, 0, stream>>>(wf2, wf2_p);

  k_stem<<<(BATCH * 196 + 255) / 256, 256, 0, stream>>>(x, w1, b1, g1, bt1, m1, v1, hpad);

  k_bconv<1, 1, 0, 1><<<6272, 256, 0, stream>>>(hpad, (const v8i*)wa_p, ba, ag, abt, am, av, fc1in, 0);
  k_bconv<3, 3, 1, 3><<<6272, 256, 0, stream>>>(hpad, (const v8i*)wb_p, bb, bg, bbt, bm, bv, fc1in, 32);
  k_bconv<5, 5, 2, 7><<<6272, 256, 0, stream>>>(hpad, (const v8i*)wc_p, bc, cg, cbt, cm, cv, fc1in, 64);

  k_fc1<<<512, 256, 0, stream>>>(fc1in, (const v8i*)wf1_p, bf1, f1g, f1bt, f1m, f1v, fc1out);

  k_fc2<<<(BATCH + 255) / 256, 256, 0, stream>>>(fc1out, wf2_p, bf2, f2g, f2bt, f2m, f2v,
                                                 (float*)d_out);
}
